// ThoughtAttention_1151051236063
// MI455X (gfx1250) — matrix operand order repaired and checked by reading, not compiled, not run on hardware
//
#include <hip/hip_runtime.h>
#include <hip/hip_bf16.h>

typedef __bf16 bf16;
typedef __attribute__((ext_vector_type(16))) __bf16 v16bf;
typedef __attribute__((ext_vector_type(8)))  __bf16 bf16x8;
typedef __attribute__((ext_vector_type(8)))  float  v8f;

// Problem constants (fixed by the harness)
#define BB   64
#define NN   196      // 14*14
#define DD   768
#define NH   12
#define DH   64
#define MM   (BB*NN)  // 12544
#define NP   208      // N padded to 13*16 (row tiles)
#define KP   224      // key dim padded to 7*32 (WMMA K chunks)

// ---------------- WMMA fragment helpers (CDNA5 16x16x32 bf16) -------------
// A fragment (16x32, row major source):
//   lanes 0-15 : row M=lane,    K=0..15 contiguous
//   lanes16-31 : row M=lane-16, K=16..31 contiguous
__device__ __forceinline__ v16bf load_a_frag(const bf16* tile, int stride, int lane) {
  int r  = lane & 15;
  int kb = (lane & 16) ? 16 : 0;
  const bf16* p = tile + r * stride + kb;
  union { v16bf v; bf16x8 h[2]; } u;
  u.h[0] = *(const bf16x8*)(p);
  u.h[1] = *(const bf16x8*)(p + 8);
  return u.v;
}
// B fragment (32x16). Source stored row-major [n][k] (K contiguous per col):
//   lanes 0-15 : col N=lane,    K=0..15 contiguous
//   lanes16-31 : col N=lane-16, K=16..31 contiguous
__device__ __forceinline__ v16bf load_b_frag(const bf16* tile, int stride, int lane) {
  int n  = lane & 15;
  int kb = (lane & 16) ? 16 : 0;
  return *(const v16bf*)(tile + n * stride + kb);
}
__device__ __forceinline__ v8f wmma_bf16(v16bf a, v16bf b, v8f c) {
  return __builtin_amdgcn_wmma_f32_16x16x32_bf16(false, a, false, b, (short)0, c,
                                                 false, false);
}

// ---------------- tiny utility kernels ------------------------------------
__global__ void k_cvt(const float* __restrict__ x, bf16* __restrict__ y, int n) {
  int i = blockIdx.x * blockDim.x + threadIdx.x;
  int s = gridDim.x * blockDim.x;
  for (; i < n; i += s) y[i] = (bf16)x[i];
}
__global__ void k_zero(bf16* __restrict__ y, int n) {
  int i = blockIdx.x * blockDim.x + threadIdx.x;
  int s = gridDim.x * blockDim.x;
  for (; i < n; i += s) y[i] = (bf16)0.f;
}

// ---------------- QKV projections: Y = X @ W^T, fused reshape -------------
// 128x128 block tile, 8 waves: wave w owns rows w*16..w*16+15 x all 128 cols.
// Double-buffered LDS W tile: 1 barrier / 8 WMMAs, staging overlaps compute.
// grid (98, 6, 3): z = mode (0:Q 1:K 2:V)
__global__ void __launch_bounds__(256)
k_proj(const bf16* __restrict__ Xq, const bf16* __restrict__ Xk,
       const bf16* __restrict__ Xv, const bf16* __restrict__ Wq,
       const bf16* __restrict__ Wk, const bf16* __restrict__ Wv,
       bf16* __restrict__ Qo, bf16* __restrict__ Ko, bf16* __restrict__ Vo) {
  __shared__ __align__(64) bf16 Wt[2][128][32];   // 16 KB double buffer
  const int mode = blockIdx.z;
  const bf16* X = (mode == 0) ? Xq : (mode == 1) ? Xk : Xv;
  const bf16* W = (mode == 0) ? Wq : (mode == 1) ? Wk : Wv;
  const int m0 = blockIdx.x * 128;
  const int n0 = blockIdx.y * 128;
  const int tid = threadIdx.x, lane = tid & 31, wave = tid >> 5;

  v8f acc[8];
  for (int j = 0; j < 8; j++) acc[j] = (v8f){0,0,0,0,0,0,0,0};
  const bf16* Arow = X + (size_t)(m0 + wave * 16) * DD;

  // Each thread stages 16 bf16 of the 128x32 W slice per K step.
  const int sr = tid >> 1, sc = (tid & 1) * 16;
  const bf16* Wsrc = W + (size_t)(n0 + sr) * DD + sc;

  { // prologue: stage K chunk 0 into buffer 0
    bf16x8 w0 = *(const bf16x8*)(Wsrc);
    bf16x8 w1 = *(const bf16x8*)(Wsrc + 8);
    *(bf16x8*)&Wt[0][sr][sc]     = w0;
    *(bf16x8*)&Wt[0][sr][sc + 8] = w1;
  }
  __syncthreads();

  for (int kc = 0; kc < DD / 32; kc++) {
    if (kc + 1 < DD / 32) {  // stage next chunk into the other buffer
      const bf16* s = Wsrc + (kc + 1) * 32;
      bf16x8 w0 = *(const bf16x8*)(s);
      bf16x8 w1 = *(const bf16x8*)(s + 8);
      int nb = (kc + 1) & 1;
      *(bf16x8*)&Wt[nb][sr][sc]     = w0;
      *(bf16x8*)&Wt[nb][sr][sc + 8] = w1;
      __builtin_prefetch(Arow + (kc + 2) * 32, 0, 1);  // global_prefetch_b8
    }
    v16bf a = load_a_frag(Arow + kc * 32, DD, lane);
    const bf16* wb = &Wt[kc & 1][0][0];
    for (int j = 0; j < 8; j++) {
      v16bf b = load_b_frag(wb + j * 16 * 32, 32, lane);
      acc[j] = wmma_bf16(a, b, acc[j]);
    }
    __syncthreads();
  }

  // Epilogue: scatter into attention-friendly layouts
  const int roff = (lane & 16) ? 8 : 0;
  for (int j = 0; j < 8; j++) {
    int col = n0 + j * 16 + (lane & 15);
    int h = col >> 6, d = col & 63;
    for (int i = 0; i < 8; i++) {
      int m = m0 + wave * 16 + i + roff;
      int b = m / NN, n = m - b * NN;
      float v = acc[j][i];
      if (mode == 0) {        // Q: pre-scale by dh^-0.5 = 0.125 (exact in bf16)
        Qo[(((size_t)b * NH + h) * NP + n) * DH + d] = (bf16)(v * 0.125f);
      } else if (mode == 1) { // K: [b][h][NP][DH]
        Ko[(((size_t)b * NH + h) * NP + n) * DH + d] = (bf16)v;
      } else {                // V transposed: [b][h][DH][KP] (key contiguous)
        Vo[(((size_t)b * NH + h) * DH + d) * KP + n] = (bf16)v;
      }
    }
  }
}

// ---------------- attention core: S = QK^T + bias; softmax; O = P V -------
// grid (4, 12, 64): x = 64-row block, y = head, z = batch
__global__ void __launch_bounds__(256)
k_attn(const bf16* __restrict__ Q, const bf16* __restrict__ K,
       const bf16* __restrict__ Vt, const float* __restrict__ bias_table,
       bf16* __restrict__ O) {
  __shared__ __align__(64) float S[64][KP];  // 57.3 KB; P (bf16) aliases rows
  __shared__ float rinv[64];
  const int rb = blockIdx.x, h = blockIdx.y, b = blockIdx.z;
  const int rowTile0 = rb * 4;
  int nT = 13 - rowTile0; if (nT > 4) nT = 4;
  const int r0 = rowTile0 * 16;
  const int tid = threadIdx.x, lane = tid & 31, wave = tid >> 5;
  const bf16* Qb = Q  + ((size_t)b * NH + h) * NP * DH;
  const bf16* Kb = K  + ((size_t)b * NH + h) * NP * DH;
  const bf16* Vb = Vt + ((size_t)b * NH + h) * DH * KP;

  // Phase 1: score tiles (nT x 13 tiles of 16x16), K-dim = 64 -> 2 WMMAs
  const int ntile = nT * 13;
  for (int t = wave; t < ntile; t += 8) {
    int tr = t / 13, tc = t - tr * 13;
    v8f acc = (v8f){0,0,0,0,0,0,0,0};
    const bf16* qa = Qb + (r0 + tr * 16) * DH;
    const bf16* ka = Kb + (tc * 16) * DH;
    for (int k0 = 0; k0 < DH; k0 += 32) {
      v16bf a  = load_a_frag(qa + k0, DH, lane);
      v16bf bv = load_b_frag(ka + k0, DH, lane);
      acc = wmma_bf16(a, bv, acc);
    }
    int col  = tc * 16 + (lane & 15);
    int roff = (lane & 16) ? 8 : 0;
    for (int i = 0; i < 8; i++) {
      int lr = tr * 16 + i + roff;
      int q = r0 + lr, kk = col;
      float v = acc[i];
      if (q < NN && kk < NN) {  // rel-pos bias, MAX_REL=20 -> +19, *39
        int qy = q / 14,  qx = q - qy * 14;
        int ky = kk / 14, kx = kk - ky * 14;
        int idx = (qy - ky + 19) * 39 + (qx - kx + 19);
        v += bias_table[idx * NH + h];
      }
      S[lr][col] = v;
    }
  }
  __syncthreads();

  // Phase 2: row softmax (wave-parallel). Write unnormalized exp() as bf16
  // aliased over the same LDS row (bf16 write at byte 2c only clobbers floats
  // already consumed at <= the same lockstep iteration); 1/rowsum deferred.
  bf16* P = (bf16*)&S[0][0];        // row lr lives at P + lr*448
  const int nrows = nT * 16;
  for (int lr = wave; lr < nrows; lr += 8) {
    int q = r0 + lr;
    bf16* prow = P + lr * (KP * 2);
    if (q >= NN) {                   // pad rows feeding last row-tile: zero
      for (int c = lane; c < KP; c += 32) prow[c] = (bf16)0.f;
      if (lane == 0) rinv[lr] = 1.f;
      continue;
    }
    float m = -3.0e38f;
    for (int c = lane; c < NN; c += 32) m = fmaxf(m, S[lr][c]);
    for (int off = 16; off > 0; off >>= 1) m = fmaxf(m, __shfl_xor(m, off, 32));
    float sum = 0.f;
    for (int c = lane; c < KP; c += 32) {
      float e = 0.f;
      if (c < NN) e = __expf(S[lr][c] - m);
      sum += e;
      prow[c] = (bf16)e;             // cols >= 196 zero-padded for PV K chunks
    }
    for (int off = 16; off > 0; off >>= 1) sum += __shfl_xor(sum, off, 32);
    if (lane == 0) rinv[lr] = 1.f / sum;
  }
  __syncthreads();

  // Phase 3: O tiles (nT x 4), K-dim = 224 -> 7 WMMAs
  const int otile = nT * 4;
  for (int t = wave; t < otile; t += 8) {
    int tr = t >> 2, tc = t & 3;
    v8f acc = (v8f){0,0,0,0,0,0,0,0};
    const bf16* pa = P  + (tr * 16) * (KP * 2);
    const bf16* va = Vb + (tc * 16) * KP;
    for (int kc = 0; kc < 7; kc++) {
      int k0 = kc * 32;
      v16bf a  = load_a_frag(pa + k0, KP * 2, lane);
      v16bf bv = load_b_frag(va + k0, KP, lane);
      acc = wmma_bf16(a, bv, acc);
    }
    int col  = tc * 16 + (lane & 15);
    int roff = (lane & 16) ? 8 : 0;
    for (int i = 0; i < 8; i++) {
      int lr = tr * 16 + i + roff;
      int q = r0 + lr;
      if (q < NN) {
        float v = acc[i] * rinv[lr];
        O[((size_t)b * NN + q) * DD + h * DH + col] = (bf16)v;
      }
    }
  }
}

// ---------------- output projection + residual: Y = O@Wo^T + query --------
// Same 128x128 double-buffered structure as k_proj. grid (98, 6)
__global__ void __launch_bounds__(256)
k_oproj(const bf16* __restrict__ O, const bf16* __restrict__ Wo,
        const float* __restrict__ query, float* __restrict__ Y) {
  __shared__ __align__(64) bf16 Wt[2][128][32];
  const int m0 = blockIdx.x * 128;
  const int n0 = blockIdx.y * 128;
  const int tid = threadIdx.x, lane = tid & 31, wave = tid >> 5;
  v8f acc[8];
  for (int j = 0; j < 8; j++) acc[j] = (v8f){0,0,0,0,0,0,0,0};
  const bf16* Arow = O + (size_t)(m0 + wave * 16) * DD;

  const int sr = tid >> 1, sc = (tid & 1) * 16;
  const bf16* Wsrc = Wo + (size_t)(n0 + sr) * DD + sc;
  {
    bf16x8 w0 = *(const bf16x8*)(Wsrc);
    bf16x8 w1 = *(const bf16x8*)(Wsrc + 8);
    *(bf16x8*)&Wt[0][sr][sc]     = w0;
    *(bf16x8*)&Wt[0][sr][sc + 8] = w1;
  }
  __syncthreads();

  for (int kc = 0; kc < DD / 32; kc++) {
    if (kc + 1 < DD / 32) {
      const bf16* s = Wsrc + (kc + 1) * 32;
      bf16x8 w0 = *(const bf16x8*)(s);
      bf16x8 w1 = *(const bf16x8*)(s + 8);
      int nb = (kc + 1) & 1;
      *(bf16x8*)&Wt[nb][sr][sc]     = w0;
      *(bf16x8*)&Wt[nb][sr][sc + 8] = w1;
      __builtin_prefetch(Arow + (kc + 2) * 32, 0, 1);
    }
    v16bf a = load_a_frag(Arow + kc * 32, DD, lane);
    const bf16* wb = &Wt[kc & 1][0][0];
    for (int j = 0; j < 8; j++) {
      v16bf b = load_b_frag(wb + j * 16 * 32, 32, lane);
      acc[j] = wmma_bf16(a, b, acc[j]);
    }
    __syncthreads();
  }

  const int roff = (lane & 16) ? 8 : 0;
  for (int j = 0; j < 8; j++) {
    int col = n0 + j * 16 + (lane & 15);
    for (int i = 0; i < 8; i++) {
      size_t m = (size_t)(m0 + wave * 16 + i + roff);
      Y[m * DD + col] = acc[j][i] + query[m * DD + col];
    }
  }
}

// ---------------- LayerNorm (weight=1, bias=0, eps=1e-5) ------------------
__global__ void __launch_bounds__(256)
k_lnorm(const float* __restrict__ Y, float* __restrict__ out) {
  const int lane = threadIdx.x & 31, wave = threadIdx.x >> 5;
  const size_t row = (size_t)blockIdx.x * 8 + wave;
  const float* y = Y + row * DD;
  float s = 0.f, s2 = 0.f;
  for (int c = lane; c < DD; c += 32) { float v = y[c]; s += v; s2 += v * v; }
  for (int off = 16; off > 0; off >>= 1) {
    s  += __shfl_xor(s,  off, 32);
    s2 += __shfl_xor(s2, off, 32);
  }
  const float mean = s * (1.f / DD);
  const float var  = s2 * (1.f / DD) - mean * mean;
  const float inv  = rsqrtf(var + 1e-5f);
  float* o = out + row * DD;
  for (int c = lane; c < DD; c += 32) o[c] = (y[c] - mean) * inv;
}

// ---------------- host-side orchestration ---------------------------------
extern "C" void kernel_launch(void* const* d_in, const int* in_sizes, int n_in,
                              void* d_out, int out_size, void* d_ws, size_t ws_size,
                              hipStream_t stream) {
  const float* query = (const float*)d_in[0];
  const float* key   = (const float*)d_in[1];
  const float* value = (const float*)d_in[2];
  const float* WQ    = (const float*)d_in[3];
  const float* WK    = (const float*)d_in[4];
  const float* WV    = (const float*)d_in[5];
  const float* WO    = (const float*)d_in[6];
  const float* bias  = (const float*)d_in[7];

  const size_t szX = (size_t)MM * DD;          // 9,633,792 elems
  const size_t szW = (size_t)DD * DD;          // 589,824
  const size_t szQ = (size_t)BB * NH * NP * DH;// 10,223,616
  const size_t szV = (size_t)BB * NH * DH * KP;// 11,010,048

  bf16* xq = (bf16*)d_ws;
  bf16* xk = xq + szX;
  bf16* xv = xk + szX;
  bf16* wq = xv + szX;
  bf16* wk = wq + szW;
  bf16* wv = wk + szW;
  bf16* wo = wv + szW;
  bf16* Qb = wo + szW;
  bf16* Kb = Qb + szQ;
  bf16* Vt = Kb + szQ;
  bf16* Obf = xq;          // alias: xq dead after Q projection
  float* Yf = (float*)xk;  // alias: xk+xv region (exactly MM*DD floats) dead too

  k_cvt<<<2048, 256, 0, stream>>>(query, xq, (int)szX);
  k_cvt<<<2048, 256, 0, stream>>>(key,   xk, (int)szX);
  k_cvt<<<2048, 256, 0, stream>>>(value, xv, (int)szX);
  k_cvt<<<1024, 256, 0, stream>>>(WQ, wq, (int)szW);
  k_cvt<<<1024, 256, 0, stream>>>(WK, wk, (int)szW);
  k_cvt<<<1024, 256, 0, stream>>>(WV, wv, (int)szW);
  k_cvt<<<1024, 256, 0, stream>>>(WO, wo, (int)szW);
  k_zero<<<2048, 256, 0, stream>>>(Vt, (int)szV);  // zero key-pad cols for PV

  dim3 gp(MM / 128, DD / 128, 3);
  k_proj<<<gp, 256, 0, stream>>>(xq, xk, xv, wq, wk, wv, Qb, Kb, Vt);

  dim3 ga(4, NH, BB);
  k_attn<<<ga, 256, 0, stream>>>(Qb, Kb, Vt, bias, Obf);

  dim3 go(MM / 128, DD / 128);
  k_oproj<<<go, 256, 0, stream>>>(Obf, wo, query, Yf);

  k_lnorm<<<MM / 8, 256, 0, stream>>>(Yf, (float*)d_out);
}
